// HeteroGNN_57071525429876
// MI455X (gfx1250) — matrix operand order repaired and checked by reading, not compiled, not run on hardware
//
#include <hip/hip_runtime.h>
#include <math.h>

typedef __bf16 bf16_t;
typedef __attribute__((ext_vector_type(16))) __bf16 v16bf;
typedef __attribute__((ext_vector_type(8)))  __bf16 v8bf;
typedef __attribute__((ext_vector_type(8)))  float   v8f;

#define TPB 256

// ---------------------------------------------------------------- utilities

__global__ __launch_bounds__(TPB)
void k_fill_f32(float* __restrict__ p, float v, int n) {
  for (int i = blockIdx.x * blockDim.x + threadIdx.x; i < n; i += gridDim.x * blockDim.x)
    p[i] = v;
}

// transpose + convert: src[K][N] f32  ->  dst[N][K] bf16
__global__ __launch_bounds__(TPB)
void k_cvtT_bf16(const float* __restrict__ src, bf16_t* __restrict__ dst, int K, int N) {
  __shared__ float t[32][33];
  const int kb = blockIdx.y * 32, nb = blockIdx.x * 32;
  const int tx = threadIdx.x & 31, ty = threadIdx.x >> 5;   // 32 x 8
  #pragma unroll
  for (int r = ty; r < 32; r += 8) {
    int k = kb + r, n = nb + tx;
    t[r][tx] = (k < K && n < N) ? src[(size_t)k * N + n] : 0.0f;
  }
  __syncthreads();
  #pragma unroll
  for (int r = ty; r < 32; r += 8) {
    int n = nb + r, k = kb + tx;
    if (n < N && k < K) dst[(size_t)n * K + k] = (bf16_t)t[tx][r];
  }
}

__device__ __forceinline__ float waveReduceSum(float v) {
  #pragma unroll
  for (int o = 16; o > 0; o >>= 1) v += __shfl_xor(v, o, 32);
  return v;
}

__device__ __forceinline__ void atomicMaxF(float* a, float v) {
  if (v >= 0.0f) atomicMax((int*)a, __float_as_int(v));
  else           atomicMin((unsigned int*)a, __float_as_uint(v));
}

// async 16B global -> LDS copy (gfx1250, tracked by ASYNCcnt)
__device__ __forceinline__ void async_g2l_b128(unsigned lds_off, const void* gaddr) {
  asm volatile("global_load_async_to_lds_b128 %0, %1, off"
               :: "v"(lds_off), "v"(gaddr) : "memory");
}
__device__ __forceinline__ void wait_async(int pending) {
  if (pending) asm volatile("s_wait_asynccnt 0x2" ::: "memory");
  else         asm volatile("s_wait_asynccnt 0x0" ::: "memory");
}

// ---------------------------------------------------------------- WMMA GEMM
// C[M,N] = A[M,K](f32) x BT[N,K](bf16, pre-transposed)  (+bias, +relu)
// block tile 128x128x32, 256 threads = 8 waves, wave tile 32x64 (2x4 WMMA tiles)
// Requires M % 128 == 0 and N % 128 == 0 (true for every call below).

#define BM 128
#define BN 128
#define BK 32
#define LDA 40   // padded LDS row stride (elements)
#define LDB 40

__device__ __forceinline__ void gemm_loadA(const float* __restrict__ A, int K,
                                           int m0, int k0, int tid, float4* ar) {
  const int c4 = (tid & 7) * 4;
  const int r0 = tid >> 3;
  const bool fast = (k0 + BK <= K) && ((K & 3) == 0);
  #pragma unroll
  for (int it = 0; it < 4; it++) {
    const int r = r0 + 32 * it;
    const float* p = A + (size_t)(m0 + r) * K + k0 + c4;
    if (fast) {
      ar[it] = *(const float4*)p;
    } else {
      float4 v; v.x = v.y = v.z = v.w = 0.0f;
      const int rem = K - k0;
      if (c4 + 0 < rem) v.x = p[0];
      if (c4 + 1 < rem) v.y = p[1];
      if (c4 + 2 < rem) v.z = p[2];
      if (c4 + 3 < rem) v.w = p[3];
      ar[it] = v;
    }
  }
}

__device__ __forceinline__ void gemm_storeA(bf16_t* __restrict__ Asb, int tid,
                                            const float4* ar) {
  const int c4 = (tid & 7) * 4;
  const int r0 = tid >> 3;
  #pragma unroll
  for (int it = 0; it < 4; it++) {
    const int r = r0 + 32 * it;
    union { bf16_t h[4]; uint2 u; } pk;
    pk.h[0] = (bf16_t)ar[it].x; pk.h[1] = (bf16_t)ar[it].y;
    pk.h[2] = (bf16_t)ar[it].z; pk.h[3] = (bf16_t)ar[it].w;
    *(uint2*)&Asb[r * LDA + c4] = pk.u;
  }
}

// fast path: async DMA of 32 contiguous bytes (16 bf16) per thread into LDS [n][k]
__device__ __forceinline__ void gemm_issueB_async(const bf16_t* __restrict__ BT, int K,
                                                  int n0, int k0, int tid,
                                                  bf16_t* __restrict__ Bsb) {
  const int n    = tid >> 1;        // 0..127
  const int half = tid & 1;         // 16-element half of BK
  const bf16_t* g = BT + (size_t)(n0 + n) * K + k0 + half * 16;
  const unsigned l = (unsigned)(uintptr_t)&Bsb[n * LDB + half * 16];
  async_g2l_b128(l,      g);
  async_g2l_b128(l + 32, g + 8);    // +32 bytes LDS, +8 elements global
}

// fallback (K not 16B-row-aligned or partial tile): masked scalar staging
__device__ __forceinline__ void gemm_stageB_slow(const bf16_t* __restrict__ BT, int K,
                                                 int n0, int k0, int tid,
                                                 bf16_t* __restrict__ Bsb) {
  const int n    = tid >> 1;
  const int half = tid & 1;
  const bf16_t* g = BT + (size_t)(n0 + n) * K;
  #pragma unroll
  for (int j = 0; j < 16; j++) {
    const int k = k0 + half * 16 + j;
    Bsb[n * LDB + half * 16 + j] = (k < K) ? g[k] : (bf16_t)0.0f;
  }
}

__global__ __launch_bounds__(TPB)
void k_gemm_bf16_wmma(const float* __restrict__ A, const bf16_t* __restrict__ BT,
                      const float* __restrict__ bias, float* __restrict__ C,
                      int M, int N, int K, int doRelu)
{
  __shared__ __align__(16) bf16_t Asb[BM * LDA];
  __shared__ __align__(16) bf16_t Bsb[2][BN * LDB];

  const int tid  = threadIdx.x;
  const int wave = tid >> 5;
  const int lane = tid & 31;
  const int waveM = wave & 3;
  const int waveN = wave >> 2;
  const int m0 = blockIdx.x * BM;
  const int n0 = blockIdx.y * BN;
  const bool fastB = ((K & 7) == 0);   // 16B-aligned rows + whole tiles

  v8f acc[2][4];
  #pragma unroll
  for (int i = 0; i < 2; i++)
    #pragma unroll
    for (int j = 0; j < 4; j++)
      #pragma unroll
      for (int r = 0; r < 8; r++) acc[i][j][r] = 0.0f;

  const int row16 = lane & 15;
  const int kh = (lane >> 4) << 3;    // A frag k-offset: 0 or 8
  const int kb = (lane >> 4) << 3;    // B frag k-offset: 0 or 8

  float4 ar[4];
  gemm_loadA(A, K, m0, 0, tid, ar);
  if (fastB) gemm_issueB_async(BT, K, n0, 0, tid, Bsb[0]);
  else       gemm_stageB_slow(BT, K, n0, 0, tid, Bsb[0]);

  int cur = 0;
  for (int k0 = 0; k0 < K; k0 += BK) {
    gemm_storeA(Asb, tid, ar);
    const int kn = k0 + BK;
    if (fastB) {
      if (kn < K) { gemm_issueB_async(BT, K, n0, kn, tid, Bsb[cur ^ 1]); wait_async(1); }
      else        { wait_async(0); }
    } else if (kn < K) {
      gemm_stageB_slow(BT, K, n0, kn, tid, Bsb[cur ^ 1]);
    }
    __syncthreads();

    if (kn < K) gemm_loadA(A, K, m0, kn, tid, ar);   // overlap with frags+WMMA

    const bf16_t* Bb = Bsb[cur];
    v16bf afrag[2], bfrag[4];
    #pragma unroll
    for (int i = 0; i < 2; i++) {
      const bf16_t* p = &Asb[(waveM * 32 + i * 16 + row16) * LDA + kh];
      union { v16bf v; v8bf h[2]; } u;
      u.h[0] = *(const v8bf*)(p);
      u.h[1] = *(const v8bf*)(p + 16);
      afrag[i] = u.v;
    }
    #pragma unroll
    for (int j = 0; j < 4; j++) {
      const bf16_t* p = &Bb[(waveN * 64 + j * 16 + row16) * LDB + kb];
      union { v16bf v; v8bf h[2]; } u;
      u.h[0] = *(const v8bf*)(p);
      u.h[1] = *(const v8bf*)(p + 16);
      bfrag[j] = u.v;
    }
    #pragma unroll
    for (int i = 0; i < 2; i++)
      #pragma unroll
      for (int j = 0; j < 4; j++)
        acc[i][j] = __builtin_amdgcn_wmma_f32_16x16x32_bf16(
            false, afrag[i], false, bfrag[j], (short)0, acc[i][j], false, false);
    __syncthreads();
    cur ^= 1;
  }

  // epilogue: C/D layout — VGPR r: lanes 0-15 row=r, lanes 16-31 row=r+8
  const int rowAdd = (lane >> 4) << 3;
  const int colL = lane & 15;
  #pragma unroll
  for (int i = 0; i < 2; i++) {
    const int rbase = m0 + waveM * 32 + i * 16 + rowAdd;
    #pragma unroll
    for (int j = 0; j < 4; j++) {
      const int col = n0 + waveN * 64 + j * 16 + colL;
      const float bb = bias ? bias[col] : 0.0f;
      #pragma unroll
      for (int r = 0; r < 8; r++) {
        float v = acc[i][j][r] + bb;
        if (doRelu) v = fmaxf(v, 0.0f);
        C[(size_t)(rbase + r) * N + col] = v;
      }
    }
  }
}

// ---------------------------------------------------------------- attention

__global__ __launch_bounds__(TPB)
void k_rowdot2(const float* __restrict__ hs, const float* __restrict__ hd,
               const float* __restrict__ as, const float* __restrict__ ad,
               float* __restrict__ es, float* __restrict__ ed, int M, int C) {
  int w = (blockIdx.x * blockDim.x + threadIdx.x) >> 5;
  int lane = threadIdx.x & 31;
  if (w >= M) return;
  const float* h1 = hs + (size_t)w * C;
  const float* h2 = hd + (size_t)w * C;
  float s1 = 0.0f, s2 = 0.0f;
  for (int c = lane; c < C; c += 32) { s1 += h1[c] * as[c]; s2 += h2[c] * ad[c]; }
  s1 = waveReduceSum(s1); s2 = waveReduceSum(s2);
  if (lane == 0) { es[w] = s1; ed[w] = s2; }
}

__global__ __launch_bounds__(TPB)
void k_edge_scores(const int* __restrict__ ei, int E0, int Etot,
                   const float* __restrict__ es, const float* __restrict__ ed,
                   float* __restrict__ ebuf, float* __restrict__ m) {
  int e = blockIdx.x * blockDim.x + threadIdx.x;
  if (e >= Etot) return;
  int src, dst;
  if (e < E0) { src = ei[e]; dst = ei[E0 + e]; } else { src = dst = e - E0; }
  float x = es[src] + ed[dst];
  x = (x >= 0.0f) ? x : 0.2f * x;          // leaky_relu 0.2
  ebuf[e] = x;
  atomicMaxF(&m[dst], x);
}

__global__ __launch_bounds__(TPB)
void k_edge_exp(const int* __restrict__ ei, int E0, int Etot,
                const float* __restrict__ ebuf, const float* __restrict__ m,
                float* __restrict__ pbuf, float* __restrict__ ssum) {
  int e = blockIdx.x * blockDim.x + threadIdx.x;
  if (e >= Etot) return;
  int dst = (e < E0) ? ei[E0 + e] : (e - E0);
  float p = expf(ebuf[e] - m[dst]);
  pbuf[e] = p;
  atomicAdd(&ssum[dst], p);
}

__global__ __launch_bounds__(TPB)
void k_edge_scatter(const int* __restrict__ ei, int E0, int Etot,
                    const float* __restrict__ hs, const float* __restrict__ pbuf,
                    const float* __restrict__ ssum, float* __restrict__ out, int C) {
  int e = blockIdx.x;
  if (e >= Etot) return;
  int src, dst;
  if (e < E0) { src = ei[e]; dst = ei[E0 + e]; } else { src = dst = e - E0; }
  float alpha = pbuf[e] / (ssum[dst] + 1e-16f);
  const float* hrow = hs + (size_t)src * C;
  float* orow = out + (size_t)dst * C;
  for (int c = threadIdx.x; c < C; c += blockDim.x)
    atomicAdd(&orow[c], hrow[c] * alpha);
}

__global__ __launch_bounds__(TPB)
void k_bias_relu_rows(float* __restrict__ x, const float* __restrict__ b, int M, int N) {
  int row = blockIdx.x;
  if (row >= M) return;
  float* p = x + (size_t)row * N;
  for (int c = threadIdx.x; c < N; c += blockDim.x)
    p[c] = fmaxf(p[c] + b[c], 0.0f);
}

// ---------------------------------------------------------------- norms / fusion

__global__ __launch_bounds__(TPB)
void k_l2norm_rows(const float* __restrict__ src, float* __restrict__ dst, int M, int N) {
  int row = blockIdx.x;
  if (row >= M) return;
  __shared__ float red[8];
  const float* s = src + (size_t)row * N;
  float ss = 0.0f;
  for (int c = threadIdx.x; c < N; c += blockDim.x) { float v = s[c]; ss += v * v; }
  ss = waveReduceSum(ss);
  int w = threadIdx.x >> 5, lane = threadIdx.x & 31;
  if (lane == 0) red[w] = ss;
  __syncthreads();
  if (threadIdx.x == 0) {
    float t = 0.0f;
    for (int i = 0; i < TPB / 32; i++) t += red[i];
    red[0] = 1.0f / fmaxf(sqrtf(t), 1e-12f);
  }
  __syncthreads();
  float sc = red[0];
  float* d = dst + (size_t)row * N;
  for (int c = threadIdx.x; c < N; c += blockDim.x) d[c] = s[c] * sc;
}

__global__ __launch_bounds__(TPB)
void k_build_hidden(const float* __restrict__ drug, const float* __restrict__ cell3,
                    const int* __restrict__ d1, const int* __restrict__ d2,
                    float* __restrict__ hidden, int Bsz, int C) {
  int b = blockIdx.x;
  if (b >= Bsz) return;
  __shared__ float red[8];
  const float* s1 = drug + (size_t)d1[b] * C;
  const float* s2 = drug + (size_t)d2[b] * C;
  const float* s3 = cell3 + (size_t)b * (2 * C);
  float ss = 0.0f;
  for (int c = threadIdx.x; c < C; c += blockDim.x)     { float v = s1[c]; ss += v * v; }
  for (int c = threadIdx.x; c < C; c += blockDim.x)     { float v = s2[c]; ss += v * v; }
  for (int c = threadIdx.x; c < 2 * C; c += blockDim.x) { float v = s3[c]; ss += v * v; }
  ss = waveReduceSum(ss);
  int w = threadIdx.x >> 5, lane = threadIdx.x & 31;
  if (lane == 0) red[w] = ss;
  __syncthreads();
  if (threadIdx.x == 0) {
    float t = 0.0f;
    for (int i = 0; i < TPB / 32; i++) t += red[i];
    red[0] = 1.0f / fmaxf(sqrtf(t), 1e-12f);
  }
  __syncthreads();
  float sc = red[0];
  float* d = hidden + (size_t)b * (4 * C);
  for (int c = threadIdx.x; c < C; c += blockDim.x)     d[c]         = s1[c] * sc;
  for (int c = threadIdx.x; c < C; c += blockDim.x)     d[C + c]     = s2[c] * sc;
  for (int c = threadIdx.x; c < 2 * C; c += blockDim.x) d[2 * C + c] = s3[c] * sc;
}

__global__ __launch_bounds__(TPB)
void k_cls(const float* __restrict__ h3, const float* __restrict__ w,
           const float* __restrict__ b, float* __restrict__ out, int Bsz, int K) {
  int idx = blockIdx.x * blockDim.x + threadIdx.x;
  if (idx >= Bsz * 2) return;
  int bb = idx >> 1, j = idx & 1;
  float acc = b[j];
  const float* h = h3 + (size_t)bb * K;
  for (int k = 0; k < K; k++) acc += h[k] * w[k * 2 + j];
  out[idx] = acc;
}

// ---------------------------------------------------------------- host launch

static inline int cdiv(int a, int b) { return (a + b - 1) / b; }

extern "C" void kernel_launch(void* const* d_in, const int* in_sizes, int n_in,
                              void* d_out, int out_size, void* d_ws, size_t ws_size,
                              hipStream_t stream) {
  (void)in_sizes; (void)n_in; (void)out_size; (void)ws_size;

  const int ND = 16384, Bsz = 4096, C = 768, FD = 256;
  const int E0 = 131072, ET = E0 + ND;

  const float* x_drug  = (const float*)d_in[0];
  const float* cellf   = (const float*)d_in[2];
  const float* Wsrc    = (const float*)d_in[3];
  const float* Wdst    = (const float*)d_in[4];
  const float* a_src   = (const float*)d_in[5];
  const float* a_dst   = (const float*)d_in[6];
  const float* b_dd    = (const float*)d_in[7];
  const float* red1_w  = (const float*)d_in[18]; const float* red1_b = (const float*)d_in[19];
  const float* red2_w  = (const float*)d_in[20]; const float* red2_b = (const float*)d_in[21];
  const float* red3_w  = (const float*)d_in[22]; const float* red3_b = (const float*)d_in[23];
  const float* r2a_w   = (const float*)d_in[24]; const float* r2a_b  = (const float*)d_in[25];
  const float* r2b_w   = (const float*)d_in[26]; const float* r2b_b  = (const float*)d_in[27];
  const float* r2c_w   = (const float*)d_in[28]; const float* r2c_b  = (const float*)d_in[29];
  const float* cls_w   = (const float*)d_in[30]; const float* cls_b  = (const float*)d_in[31];
  const int*   d1      = (const int*)d_in[32];
  const int*   d2      = (const int*)d_in[33];
  const int*   ei_dd   = (const int*)d_in[34];
  float* out           = (float*)d_out;

  // workspace layout (linear allocator, 256B aligned)
  size_t off = 0;
  char* base = (char*)d_ws;
  auto alloc = [&](size_t bytes) -> char* {
    char* p = base + off;
    off += (bytes + 255) & ~(size_t)255;
    return p;
  };
  bf16_t* wsrc_bf = (bf16_t*)alloc((size_t)FD * C * 2);
  bf16_t* wdst_bf = (bf16_t*)alloc((size_t)FD * C * 2);
  bf16_t* red1_bf = (bf16_t*)alloc((size_t)890 * 2048 * 2);
  bf16_t* red2_bf = (bf16_t*)alloc((size_t)2048 * 512 * 2);
  bf16_t* red3_bf = (bf16_t*)alloc((size_t)512 * 1536 * 2);
  bf16_t* r2a_bf  = (bf16_t*)alloc((size_t)3072 * 2048 * 2);
  bf16_t* r2b_bf  = (bf16_t*)alloc((size_t)2048 * 512 * 2);
  bf16_t* r2c_bf  = (bf16_t*)alloc((size_t)512 * 768 * 2);
  float* hs    = (float*)alloc((size_t)ND * C * 4);
  float* hdout = (float*)alloc((size_t)ND * C * 4);   // hd, then reused as aggregation/out
  float* es    = (float*)alloc((size_t)ND * 4);
  float* ed    = (float*)alloc((size_t)ND * 4);
  float* mbuf  = (float*)alloc((size_t)ND * 4);
  float* sbuf  = (float*)alloc((size_t)ND * 4);
  float* ebuf  = (float*)alloc((size_t)ET * 4);
  float* pbuf  = (float*)alloc((size_t)ET * 4);
  float* cellnorm = (float*)alloc((size_t)Bsz * 890 * 4);
  float* cell1 = (float*)alloc((size_t)Bsz * 2048 * 4);
  float* cell2 = (float*)alloc((size_t)Bsz * 512 * 4);
  float* cell3 = (float*)alloc((size_t)Bsz * 1536 * 4);
  float* hidden = (float*)alloc((size_t)Bsz * 3072 * 4);
  float* h1    = (float*)alloc((size_t)Bsz * 2048 * 4);
  float* h2    = (float*)alloc((size_t)Bsz * 512 * 4);
  float* h3    = (float*)alloc((size_t)Bsz * 768 * 4);

  // 1) weights -> bf16, transposed to [N][K] for contiguous-K staging
  struct { const float* s; bf16_t* d; int K, N; } cv[] = {
    {Wsrc,   wsrc_bf, FD,   C},    {Wdst,   wdst_bf, FD,   C},
    {red1_w, red1_bf, 890,  2048}, {red2_w, red2_bf, 2048, 512},
    {red3_w, red3_bf, 512,  1536}, {r2a_w,  r2a_bf,  3072, 2048},
    {r2b_w,  r2b_bf,  2048, 512},  {r2c_w,  r2c_bf,  512,  768},
  };
  for (auto& c : cv) {
    dim3 g(cdiv(c.N, 32), cdiv(c.K, 32));
    k_cvtT_bf16<<<g, TPB, 0, stream>>>(c.s, c.d, c.K, c.N);
  }

  // 2) hs = x_drug @ Wsrc ; hd = x_drug @ Wdst (no bias/relu)
  {
    dim3 g(cdiv(ND, BM), cdiv(C, BN));
    k_gemm_bf16_wmma<<<g, TPB, 0, stream>>>(x_drug, wsrc_bf, nullptr, hs, ND, C, FD, 0);
    k_gemm_bf16_wmma<<<g, TPB, 0, stream>>>(x_drug, wdst_bf, nullptr, hdout, ND, C, FD, 0);
  }

  // 3) per-node attention logits
  k_rowdot2<<<cdiv(ND * 32, TPB), TPB, 0, stream>>>(hs, hdout, a_src, a_dst, es, ed, ND, C);

  // 4) softmax over incoming edges, then weighted scatter into hdout (reused)
  k_fill_f32<<<4096, TPB, 0, stream>>>(hdout, 0.0f, ND * C);
  k_fill_f32<<<64, TPB, 0, stream>>>(mbuf, -INFINITY, ND);
  k_fill_f32<<<64, TPB, 0, stream>>>(sbuf, 0.0f, ND);
  k_edge_scores<<<cdiv(ET, TPB), TPB, 0, stream>>>(ei_dd, E0, ET, es, ed, ebuf, mbuf);
  k_edge_exp<<<cdiv(ET, TPB), TPB, 0, stream>>>(ei_dd, E0, ET, ebuf, mbuf, pbuf, sbuf);
  k_edge_scatter<<<ET, TPB, 0, stream>>>(ei_dd, E0, ET, hs, pbuf, sbuf, hdout, C);
  k_bias_relu_rows<<<ND, TPB, 0, stream>>>(hdout, b_dd, ND, C);  // hdout = drug embeddings

  // 5) cell MLP
  k_l2norm_rows<<<Bsz, TPB, 0, stream>>>(cellf, cellnorm, Bsz, 890);
  {
    dim3 g1(cdiv(Bsz, BM), cdiv(2048, BN));
    k_gemm_bf16_wmma<<<g1, TPB, 0, stream>>>(cellnorm, red1_bf, red1_b, cell1, Bsz, 2048, 890, 1);
    dim3 g2(cdiv(Bsz, BM), cdiv(512, BN));
    k_gemm_bf16_wmma<<<g2, TPB, 0, stream>>>(cell1, red2_bf, red2_b, cell2, Bsz, 512, 2048, 1);
    dim3 g3(cdiv(Bsz, BM), cdiv(1536, BN));
    k_gemm_bf16_wmma<<<g3, TPB, 0, stream>>>(cell2, red3_bf, red3_b, cell3, Bsz, 1536, 512, 1);
  }

  // 6) hidden = l2norm(concat[drug[d1], drug[d2], cell3])
  k_build_hidden<<<Bsz, TPB, 0, stream>>>(hdout, cell3, d1, d2, hidden, Bsz, C);

  // 7) fusion MLP
  {
    dim3 g1(cdiv(Bsz, BM), cdiv(2048, BN));
    k_gemm_bf16_wmma<<<g1, TPB, 0, stream>>>(hidden, r2a_bf, r2a_b, h1, Bsz, 2048, 3072, 1);
    dim3 g2(cdiv(Bsz, BM), cdiv(512, BN));
    k_gemm_bf16_wmma<<<g2, TPB, 0, stream>>>(h1, r2b_bf, r2b_b, h2, Bsz, 512, 2048, 1);
    dim3 g3(cdiv(Bsz, BM), cdiv(768, BN));
    k_gemm_bf16_wmma<<<g3, TPB, 0, stream>>>(h2, r2c_bf, r2c_b, h3, Bsz, 768, 512, 1);
  }

  // 8) classifier (N=2, plain dot)
  k_cls<<<cdiv(Bsz * 2, TPB), TPB, 0, stream>>>(h3, cls_w, cls_b, out, Bsz, C);
}
